// VisionAttention_17532056502551
// MI455X (gfx1250) — compile-verified, matrix operand order checked by reading, not run on hardware
//
#include <hip/hip_runtime.h>
#include <hip/hip_bf16.h>

#define HIDDEN   1280
#define HEADS    16
#define HEAD_DIM 80
#define BB       2
#define NN       2048
#define MROWS    (BB * NN)          // 4096
#define QKV_COLS (3 * HIDDEN)       // 3840

typedef __attribute__((ext_vector_type(16))) _Float16 v16h;
typedef __attribute__((ext_vector_type(8)))  _Float16 v8h;
typedef __attribute__((ext_vector_type(8)))  float    v8f;
typedef __attribute__((ext_vector_type(4)))  int      v4i;

#if __has_builtin(__builtin_amdgcn_global_load_async_to_lds_b128)
#define HAVE_ASYNC_LDS 1
#endif

__device__ __forceinline__ void wait_asynccnt0() {
#if defined(HAVE_ASYNC_LDS)
#if __has_builtin(__builtin_amdgcn_s_wait_asynccnt)
    __builtin_amdgcn_s_wait_asynccnt(0);
#else
    asm volatile("s_wait_asynccnt 0x0" ::: "memory");
#endif
#endif
}

__device__ __forceinline__ v8f wmma_f16(v16h a, v16h b, v8f c) {
    // D = A(16x32 f16) * B(32x16 f16) + C(16x16 f32)
    return __builtin_amdgcn_wmma_f32_16x16x32_f16(
        /*neg_a=*/false, a, /*neg_b=*/false, b,
        /*c_mod=*/(short)0, c, /*reuse_a=*/false, /*reuse_b=*/false);
}

// A-fragment 16x32 f16 per-lane K index pattern (ISA 7.12.2)
__device__ __forceinline__ int apat(int e, int hl) {
    return (e < 8) ? (hl * 8 + e) : (16 + hl * 8 + (e - 8));
}

__device__ __forceinline__ void store_out(float* p, float v)    { *p = v; }
__device__ __forceinline__ void store_out(_Float16* p, float v) { *p = (_Float16)v; }

struct H16 { _Float16 h[16]; };

__device__ __forceinline__ H16 load_cvt16(const float* p) {
    const float4* s = reinterpret_cast<const float4*>(p);
    H16 r;
#pragma unroll
    for (int i = 0; i < 4; ++i) {
        float4 v = s[i];
        r.h[4 * i + 0] = (_Float16)v.x;
        r.h[4 * i + 1] = (_Float16)v.y;
        r.h[4 * i + 2] = (_Float16)v.z;
        r.h[4 * i + 3] = (_Float16)v.w;
    }
    return r;
}

__device__ __forceinline__ void store_h16(_Float16* dst, const H16& r) {
    // dst is 16B aligned (callers guarantee); merge into two 16B LDS stores
    v8h lo, hi;
#pragma unroll
    for (int i = 0; i < 8; ++i) { lo[i] = r.h[i]; hi[i] = r.h[8 + i]; }
    *reinterpret_cast<v8h*>(dst)     = lo;
    *reinterpret_cast<v8h*>(dst + 8) = hi;
}

// ---------------------------------------------------------------------------
// Tiled WMMA GEMM: C[M,Ncols] = A[M,K](f32) * W[K,Ncols](f32) + bias
// Block tile 128x128, k-step 32, 256 threads = 8 waves (4x2 grid),
// wave tile 32x64 -> 8 WMMA per wave per k-step. Double-buffered LDS +
// global_prefetch two tiles ahead. (Register staging path kept because the
// f32 -> f16 conversion must happen in flight; async LDS copies are raw.)
// ---------------------------------------------------------------------------
template <typename TOUT>
__global__ __launch_bounds__(256)
void gemm_wmma_kernel(const float* __restrict__ A, const float* __restrict__ W,
                      const float* __restrict__ bias, TOUT* __restrict__ C,
                      int M, int Ncols, int K) {
    constexpr int BM = 128, BN = 128, BK = 32, PAD = 8;
    __shared__ _Float16 As[2][BM][BK + PAD];   // 2*128*40*2 = 20.0 KB
    __shared__ _Float16 Bs[2][BK][BN + PAD];   // 2*32*136*2 = 17.0 KB

    const int tid  = threadIdx.x;
    const int lane = tid & 31;
    const int wave = tid >> 5;
    const int wm   = wave & 3;     // 0..3 -> 32-row slice
    const int wn   = wave >> 2;    // 0..1 -> 64-col slice
    const int hl   = lane >> 4;
    const int lcol = lane & 15;

    const int rowBase = blockIdx.y * BM;
    const int colBase = blockIdx.x * BN;

    // staging coordinates: 16 contiguous floats per thread per tile
    const int ar = tid >> 1, ac = (tid & 1) * 16;   // A: 128 rows x 32
    const int br = tid >> 3, bc = (tid & 7) * 16;   // B: 32 rows x 128

    const float* aSrc = &A[(size_t)(rowBase + ar) * K + ac];
    const float* bSrc = &W[(size_t)br * Ncols + colBase + bc];

    v8f acc[2][4];
#pragma unroll
    for (int s = 0; s < 2; ++s)
#pragma unroll
        for (int b = 0; b < 4; ++b) acc[s][b] = (v8f){};

    const int NK = K / BK;

    // prologue: stage tile 0 into buffer 0
    {
        H16 ra = load_cvt16(aSrc);
        H16 rb = load_cvt16(bSrc);
        store_h16(&As[0][ar][ac], ra);
        store_h16(&Bs[0][br][bc], rb);
    }
    __syncthreads();

    for (int i = 0; i < NK; ++i) {
        const int buf = i & 1;
        H16 ra, rb;
        if (i + 1 < NK) {
            ra = load_cvt16(aSrc + (i + 1) * BK);
            rb = load_cvt16(bSrc + (size_t)(i + 1) * BK * Ncols);
        }
        if (i + 2 < NK) {
            __builtin_prefetch(aSrc + (i + 2) * BK, 0, 0);
            __builtin_prefetch(bSrc + (size_t)(i + 2) * BK * Ncols, 0, 0);
        }

        // build A fragments (2 x 16 rows) and sweep 4 B fragments
        v16h af[2];
#pragma unroll
        for (int s = 0; s < 2; ++s) {
#pragma unroll
            for (int e = 0; e < 16; ++e)
                af[s][e] = As[buf][wm * 32 + s * 16 + lcol][apat(e, hl)];
        }
#pragma unroll
        for (int bi = 0; bi < 4; ++bi) {
            v16h bf;
#pragma unroll
            for (int e = 0; e < 16; ++e)
                bf[e] = Bs[buf][apat(e, hl)][wn * 64 + bi * 16 + lcol];
            acc[0][bi] = wmma_f16(af[0], bf, acc[0][bi]);
            acc[1][bi] = wmma_f16(af[1], bf, acc[1][bi]);
        }

        if (i + 1 < NK) {
            store_h16(&As[buf ^ 1][ar][ac], ra);
            store_h16(&Bs[buf ^ 1][br][bc], rb);
        }
        __syncthreads();
    }

    // epilogue: bias + store
#pragma unroll
    for (int s = 0; s < 2; ++s) {
#pragma unroll
        for (int bi = 0; bi < 4; ++bi) {
            const int n  = colBase + wn * 64 + bi * 16 + lcol;
            const float bv = bias[n];
#pragma unroll
            for (int r = 0; r < 8; ++r) {
                const int m = rowBase + wm * 32 + s * 16 + r + 8 * hl;
                store_out(&C[(size_t)m * Ncols + n], acc[s][bi][r] + bv);
            }
        }
    }
}

// ---------------------------------------------------------------------------
// RoPE + head split: qkv f16 [B,N,3*HIDDEN] -> Q,K,V f16 [B,H,N,D]
// scale 1/sqrt(D) folded into Q.
// ---------------------------------------------------------------------------
__global__ __launch_bounds__(256)
void rope_split_kernel(const _Float16* __restrict__ qkv,
                       const float* __restrict__ cosb, const float* __restrict__ sinb,
                       _Float16* __restrict__ Qo, _Float16* __restrict__ Ko,
                       _Float16* __restrict__ Vo) {
    const int idx = blockIdx.x * blockDim.x + threadIdx.x;
    if (idx >= BB * NN * HEADS * HEAD_DIM) return;
    const int d = idx % HEAD_DIM;
    const int h = (idx / HEAD_DIM) % HEADS;
    const int n = (idx / (HEAD_DIM * HEADS)) % NN;
    const int b = idx / (HEAD_DIM * HEADS * NN);

    const size_t rowoff = ((size_t)b * NN + n) * QKV_COLS;
    const int hd = h * HEAD_DIM + d;
    const float c = cosb[n * HEAD_DIM + d];
    const float s = sinb[n * HEAD_DIM + d];

    const int   dro = (d < HEAD_DIM / 2) ? (d + HEAD_DIM / 2) : (d - HEAD_DIM / 2);
    const float sgn = (d < HEAD_DIM / 2) ? -1.0f : 1.0f;

    const float q  = (float)qkv[rowoff + hd];
    const float k  = (float)qkv[rowoff + HIDDEN + hd];
    const float v  = (float)qkv[rowoff + 2 * HIDDEN + hd];
    const float qr = (float)qkv[rowoff + h * HEAD_DIM + dro];
    const float kr = (float)qkv[rowoff + HIDDEN + h * HEAD_DIM + dro];

    const float scale = 0.11180339887498949f;  // 1/sqrt(80)
    const size_t o = (((size_t)b * HEADS + h) * NN + n) * HEAD_DIM + d;
    Qo[o] = (_Float16)((q * c + sgn * qr * s) * scale);
    Ko[o] = (_Float16)(k * c + sgn * kr * s);
    Vo[o] = (_Float16)v;
}

// ---------------------------------------------------------------------------
// Flash attention: block = 64 q rows x one (b,h); 4 waves x 16 q rows.
// Keys processed 64 at a time: 12 score WMMAs (D padded to 96) + online
// softmax (one butterfly per row per 64 keys) + P via LDS + 10 PV WMMAs.
// K/V tiles staged with GLOBAL_LOAD_ASYNC_TO_LDS_B128 when available.
// ---------------------------------------------------------------------------
__global__ __launch_bounds__(128)
void attn_kernel(const _Float16* __restrict__ Q, const _Float16* __restrict__ Kc,
                 const _Float16* __restrict__ Vc, float* __restrict__ O) {
    constexpr int BQ = 64, BKV = 64, DPAD = 96;
    __shared__ _Float16 Kt[BKV][DPAD + 8];       // row stride 208 B (16B-aligned)
    __shared__ _Float16 Vt[BKV][HEAD_DIM + 8];   // row stride 176 B (16B-aligned)
    __shared__ _Float16 Ps[4][16][BKV + 8];      // per-wave P tile 16x64

    const int tid  = threadIdx.x;
    const int lane = tid & 31;
    const int wave = tid >> 5;
    const int hl   = lane >> 4;
    const int lcol = lane & 15;

    const int h = blockIdx.y;
    const int b = blockIdx.z;
    const size_t bh = (size_t)b * HEADS + h;

    // zero the D-padding region of Kt once (stores below never touch d >= 80)
    for (int i = tid; i < BKV * (DPAD + 8); i += 128) {
        const int r = i / (DPAD + 8), c = i % (DPAD + 8);
        if (c >= HEAD_DIM) Kt[r][c] = (_Float16)0.0f;
    }

    // load this wave's Q fragments: 16 rows x 96 (zero-padded past 80)
    const int qrow = blockIdx.x * BQ + wave * 16 + lcol;
    const _Float16* qptr = Q + (bh * NN + qrow) * HEAD_DIM;
    v16h qf[3];
#pragma unroll
    for (int f = 0; f < 3; ++f) {
#pragma unroll
        for (int e = 0; e < 16; ++e) {
            const int k = f * 32 + apat(e, hl);
            qf[f][e] = (k < HEAD_DIM) ? qptr[k] : (_Float16)0.0f;
        }
    }

    v8f o_acc[5] = {{}, {}, {}, {}, {}};
    float mrow[8], lrow[8];
#pragma unroll
    for (int r = 0; r < 8; ++r) { mrow[r] = -3.0e38f; lrow[r] = 0.0f; }

    const char* KbaseB = reinterpret_cast<const char*>(Kc + bh * NN * HEAD_DIM);
    const char* VbaseB = reinterpret_cast<const char*>(Vc + bh * NN * HEAD_DIM);

    for (int kt0 = 0; kt0 < NN; kt0 += BKV) {
        __syncthreads();  // protect previous iteration's Kt/Vt reads
#if defined(HAVE_ASYNC_LDS)
        // async stage: 64 rows x 10 x 16B segments, 5 per thread per tensor
        for (int i = tid; i < BKV * 10; i += 128) {
            const int r = i / 10, sg = i % 10;
            const char* gk = KbaseB + ((size_t)(kt0 + r) * HEAD_DIM + sg * 8) * 2;
            const char* gv = VbaseB + ((size_t)(kt0 + r) * HEAD_DIM + sg * 8) * 2;
            __builtin_amdgcn_global_load_async_to_lds_b128(
                (v4i*)gk, (v4i*)&Kt[r][sg * 8], 0, 0);
            __builtin_amdgcn_global_load_async_to_lds_b128(
                (v4i*)gv, (v4i*)&Vt[r][sg * 8], 0, 0);
        }
        wait_asynccnt0();
#else
        // fallback stage: 64 rows x 40 dwords (80 halves), 20 dwords/thread
        const uint32_t* Ksrc = reinterpret_cast<const uint32_t*>(KbaseB);
        const uint32_t* Vsrc = reinterpret_cast<const uint32_t*>(VbaseB);
        for (int i = tid; i < BKV * 40; i += 128) {
            const int r = i / 40, c = i % 40;
            const uint32_t kv = Ksrc[(size_t)(kt0 + r) * 40 + c];
            const uint32_t vv = Vsrc[(size_t)(kt0 + r) * 40 + c];
            *reinterpret_cast<uint32_t*>(&Kt[r][2 * c]) = kv;
            *reinterpret_cast<uint32_t*>(&Vt[r][2 * c]) = vv;
        }
#endif
        __syncthreads();

        // S = Q * K^T over padded D: four 16x16 score tiles (keys nt*16..)
        v8f sc[4] = {{}, {}, {}, {}};
#pragma unroll
        for (int f = 0; f < 3; ++f) {
#pragma unroll
            for (int nt = 0; nt < 4; ++nt) {
                v16h bf;
#pragma unroll
                for (int e = 0; e < 16; ++e)
                    bf[e] = Kt[nt * 16 + lcol][f * 32 + apat(e, hl)];
                sc[nt] = wmma_f16(qf[f], bf, sc[nt]);
            }
        }

        // online softmax: one butterfly per row per 64 keys
#pragma unroll
        for (int r = 0; r < 8; ++r) {
            float mx = fmaxf(fmaxf(sc[0][r], sc[1][r]), fmaxf(sc[2][r], sc[3][r]));
#pragma unroll
            for (int off = 1; off < 16; off <<= 1)
                mx = fmaxf(mx, __shfl_xor(mx, off, 32));
            const float mnew  = fmaxf(mrow[r], mx);
            const float alpha = __expf(mrow[r] - mnew);
            float p[4], ls = 0.0f;
#pragma unroll
            for (int nt = 0; nt < 4; ++nt) { p[nt] = __expf(sc[nt][r] - mnew); ls += p[nt]; }
#pragma unroll
            for (int off = 1; off < 16; off <<= 1)
                ls += __shfl_xor(ls, off, 32);
            lrow[r] = lrow[r] * alpha + ls;
            mrow[r] = mnew;
#pragma unroll
            for (int dt = 0; dt < 5; ++dt) o_acc[dt][r] *= alpha;
            const int prow = r + 8 * hl;
#pragma unroll
            for (int nt = 0; nt < 4; ++nt)
                Ps[wave][prow][nt * 16 + lcol] = (_Float16)p[nt];
        }
        __syncthreads();  // make P visible (C-layout -> A-layout reshuffle)

        // reload P as two A fragments (16 rows x 64 keys)
        v16h pa0, pa1;
#pragma unroll
        for (int e = 0; e < 16; ++e) {
            pa0[e] = Ps[wave][lcol][apat(e, hl)];
            pa1[e] = Ps[wave][lcol][32 + apat(e, hl)];
        }

        // O += P * V : 5 output tiles x 2 key halves
#pragma unroll
        for (int dt = 0; dt < 5; ++dt) {
            v16h vb0, vb1;
#pragma unroll
            for (int e = 0; e < 16; ++e) {
                vb0[e] = Vt[apat(e, hl)][dt * 16 + lcol];
                vb1[e] = Vt[32 + apat(e, hl)][dt * 16 + lcol];
            }
            o_acc[dt] = wmma_f16(pa0, vb0, o_acc[dt]);
            o_acc[dt] = wmma_f16(pa1, vb1, o_acc[dt]);
        }
    }

    // normalize and write attention output in [B, N, H*D] layout for proj GEMM
#pragma unroll
    for (int dt = 0; dt < 5; ++dt) {
#pragma unroll
        for (int r = 0; r < 8; ++r) {
            const int m = r + 8 * hl;
            const int n = blockIdx.x * BQ + wave * 16 + m;
            const float val = o_acc[dt][r] / lrow[r];
            O[((size_t)b * NN + n) * HIDDEN + h * HEAD_DIM + dt * 16 + lcol] = val;
        }
    }
}

// ---------------------------------------------------------------------------
extern "C" void kernel_launch(void* const* d_in, const int* in_sizes, int n_in,
                              void* d_out, int out_size, void* d_ws, size_t ws_size,
                              hipStream_t stream) {
    const float* x      = (const float*)d_in[0];
    const float* cosb   = (const float*)d_in[1];
    const float* sinb   = (const float*)d_in[2];
    const float* qkv_w  = (const float*)d_in[3];
    const float* qkv_b  = (const float*)d_in[4];
    const float* proj_w = (const float*)d_in[5];
    const float* proj_b = (const float*)d_in[6];
    float* out = (float*)d_out;

    // workspace layout
    char* ws = (char*)d_ws;
    _Float16* qkv_f16 = (_Float16*)ws;                               // 4096*3840 f16
    ws += (size_t)MROWS * QKV_COLS * sizeof(_Float16);
    _Float16* Qb = (_Float16*)ws; ws += (size_t)MROWS * HIDDEN * sizeof(_Float16);
    _Float16* Kb = (_Float16*)ws; ws += (size_t)MROWS * HIDDEN * sizeof(_Float16);
    _Float16* Vb = (_Float16*)ws; ws += (size_t)MROWS * HIDDEN * sizeof(_Float16);
    float* attn_out = (float*)ws;                                    // 4096*1280 f32

    // 1) QKV GEMM -> f16
    {
        dim3 grid(QKV_COLS / 128, MROWS / 128);
        gemm_wmma_kernel<_Float16><<<grid, 256, 0, stream>>>(
            x, qkv_w, qkv_b, qkv_f16, MROWS, QKV_COLS, HIDDEN);
    }
    // 2) RoPE + head split
    {
        const int total = BB * NN * HEADS * HEAD_DIM;
        rope_split_kernel<<<(total + 255) / 256, 256, 0, stream>>>(
            qkv_f16, cosb, sinb, Qb, Kb, Vb);
    }
    // 3) Flash attention
    {
        dim3 grid(NN / 64, HEADS, BB);
        attn_kernel<<<grid, 128, 0, stream>>>(Qb, Kb, Vb, attn_out);
    }
    // 4) Output projection -> f32 d_out
    {
        dim3 grid(HIDDEN / 128, MROWS / 128);
        gemm_wmma_kernel<float><<<grid, 256, 0, stream>>>(
            attn_out, proj_w, proj_b, out, MROWS, HIDDEN, HIDDEN);
    }
}
